// Non_local_att_61950608278115
// MI455X (gfx1250) — compile-verified, matrix operand order checked by reading, not run on hardware
//
#include <hip/hip_runtime.h>
#include <math.h>

// ---------------------------------------------------------------------------
// Problem constants (from reference)
// ---------------------------------------------------------------------------
#define BB 32
#define CC 512
#define HW 1024
#define RR 256   // C/2

typedef __bf16 bf16_t;
typedef __bf16 v16bf __attribute__((ext_vector_type(16)));
typedef float  v8f   __attribute__((ext_vector_type(8)));

union Frag16 { v16bf v; uint4 q[2]; };

// round-to-nearest-even f32 -> bf16
__device__ __forceinline__ bf16_t f2bf(float f) {
    unsigned u = __float_as_uint(f);
    u += 0x7FFFu + ((u >> 16) & 1u);
    unsigned short h = (unsigned short)(u >> 16);
    return __builtin_bit_cast(bf16_t, h);
}

// Load one wave-lane's 16 bf16 K-run for a WMMA operand.
// Per ISA 16-bit A/B layout: lane half selects kbase (0 or 8); the 16
// elements are K = {kbase..kbase+7, kbase+16..kbase+23} -> two b128 loads.
__device__ __forceinline__ v16bf load_frag(const bf16_t* __restrict__ row, int kbase) {
    Frag16 f;
    f.q[0] = *reinterpret_cast<const uint4*>(row + kbase);
    f.q[1] = *reinterpret_cast<const uint4*>(row + kbase + 16);
    return f.v;
}

__device__ __forceinline__ v8f wmma_bf16(v16bf a, v16bf b, v8f c) {
    return __builtin_amdgcn_wmma_f32_16x16x32_bf16(false, a, false, b, (short)0, c,
                                                   false, false);
}

// ---------------------------------------------------------------------------
// Kernel 0: fp32 -> bf16 weight conversion
// ---------------------------------------------------------------------------
__global__ __launch_bounds__(256) void k_cvt_bf16(const float* __restrict__ in,
                                                  bf16_t* __restrict__ out, int n) {
    int i = blockIdx.x * 256 + threadIdx.x;
    if (i < n) out[i] = f2bf(in[i]);
}

// ---------------------------------------------------------------------------
// Kernel 1: Fk[b][n][r] and Fp[b][n][r]  =  X^T * W^T + bias   (K = C)
//   A tile (x^T, K strided in global) staged through LDS as [n][k] bf16.
//   B fragments (Wk/Wp rows, K contiguous) loaded directly from global bf16.
//   One shared A fragment feeds both projections.
// grid (HW/64, R/64, B), block 256 (8 waves). Wave owns one 16-row strip,
// two 16-col tiles -> 4 accumulators (K and P for each col tile).
// ---------------------------------------------------------------------------
__global__ __launch_bounds__(256) void k_proj_kp(
    const float* __restrict__ x, const bf16_t* __restrict__ Wk,
    const bf16_t* __restrict__ Wp, const float* __restrict__ bk,
    const float* __restrict__ bp, bf16_t* __restrict__ Fk,
    bf16_t* __restrict__ Fp) {
    __shared__ __align__(16) bf16_t As[64][40];   // pad 40: conflict-free strided reads

    const int tid  = threadIdx.x;
    const int wave = tid >> 5, lane = tid & 31;
    const int l16 = lane & 15, hi = lane >> 4, kbase = hi * 8;
    const int n0 = blockIdx.x * 64, r0 = blockIdx.y * 64, b = blockIdx.z;

    const int t0 = wave * 2;
    const int row_t = t0 >> 2;        // 0..3 : 16-row strip inside the 64x64 tile
    const int colA  = t0 & 3;         // col tiles colA, colA+1

    v8f accK[2] = {v8f{}, v8f{}};
    v8f accP[2] = {v8f{}, v8f{}};

    for (int k0 = 0; k0 < CC; k0 += 32) {
        __syncthreads();
        // stage x^T tile: As[n][k] = x[b, k0+k, n0+n]  (coalesced over n)
        #pragma unroll
        for (int i = 0; i < 8; ++i) {
            int idx = tid + i * 256;          // 2048 elems
            int n = idx & 63, k = idx >> 6;
            As[n][k] = f2bf(x[((size_t)(b * CC + k0 + k)) * HW + n0 + n]);
        }
        __syncthreads();

        v16bf aF = load_frag(&As[row_t * 16 + l16][0], kbase);
        #pragma unroll
        for (int j = 0; j < 2; ++j) {
            int r = r0 + (colA + j) * 16 + l16;
            v16bf bK = load_frag(Wk + (size_t)r * CC + k0, kbase);
            v16bf bP = load_frag(Wp + (size_t)r * CC + k0, kbase);
            accK[j] = wmma_bf16(aF, bK, accK[j]);
            accP[j] = wmma_bf16(aF, bP, accP[j]);
        }
    }

    #pragma unroll
    for (int j = 0; j < 2; ++j) {
        int r = r0 + (colA + j) * 16 + l16;
        float bbk = bk[r], bbp = bp[r];
        #pragma unroll
        for (int v = 0; v < 8; ++v) {
            int n = n0 + row_t * 16 + v + hi * 8;   // C/D: lanes0-15 M=v, 16-31 M=v+8
            size_t o = ((size_t)(b * HW + n)) * RR + r;
            Fk[o] = f2bf(accK[j][v] + bbk);
            Fp[o] = f2bf(accP[j][v] + bbp);
        }
    }
}

// ---------------------------------------------------------------------------
// Kernel 2: Fv[b][c][m] = Wv * X + bv   (K = C)
//   A (Wv rows, K contiguous) direct global; B (x columns) staged via LDS.
// grid (HW/64, C/64, B), block 256.
// ---------------------------------------------------------------------------
__global__ __launch_bounds__(256) void k_proj_v(
    const float* __restrict__ x, const bf16_t* __restrict__ Wv,
    const float* __restrict__ bv, bf16_t* __restrict__ Fv) {
    __shared__ __align__(16) bf16_t Bs[64][40];   // [m][k]

    const int tid  = threadIdx.x;
    const int wave = tid >> 5, lane = tid & 31;
    const int l16 = lane & 15, hi = lane >> 4, kbase = hi * 8;
    const int m0 = blockIdx.x * 64, c0 = blockIdx.y * 64, b = blockIdx.z;

    const int t0 = wave * 2;
    const int row_t = t0 >> 2;
    const int colA  = t0 & 3;

    v8f acc[2] = {v8f{}, v8f{}};

    for (int k0 = 0; k0 < CC; k0 += 32) {
        __syncthreads();
        #pragma unroll
        for (int i = 0; i < 8; ++i) {
            int idx = tid + i * 256;
            int m = idx & 63, k = idx >> 6;
            Bs[m][k] = f2bf(x[((size_t)(b * CC + k0 + k)) * HW + m0 + m]);
        }
        __syncthreads();

        int c = c0 + row_t * 16 + l16;
        v16bf aF = load_frag(Wv + (size_t)c * CC + k0, kbase);
        #pragma unroll
        for (int j = 0; j < 2; ++j) {
            v16bf bF = load_frag(&Bs[(colA + j) * 16 + l16][0], kbase);
            acc[j] = wmma_bf16(aF, bF, acc[j]);
        }
    }

    #pragma unroll
    for (int j = 0; j < 2; ++j) {
        int m = m0 + (colA + j) * 16 + l16;
        #pragma unroll
        for (int v = 0; v < 8; ++v) {
            int c = c0 + row_t * 16 + v + hi * 8;
            Fv[((size_t)(b * CC + c)) * HW + m] = f2bf(acc[j][v] + bv[c]);
        }
    }
}

// ---------------------------------------------------------------------------
// Kernel 3: energy rows + softmax -> S[b][n][m] (bf16, row-major over m)
//   energy[n,m] = sum_r Fk[n,r] * Fp[m,r]  (both [HW][R], K contiguous ->
//   direct global WMMA fragments, no LDS staging). E strip (16x1024 f32,
//   64 KB) lives in LDS for the fused softmax; wave32 __shfl_xor reductions.
// grid (HW/16, B), block 256.
// ---------------------------------------------------------------------------
__global__ __launch_bounds__(256) void k_energy_softmax(
    const bf16_t* __restrict__ Fk, const bf16_t* __restrict__ Fp,
    bf16_t* __restrict__ S) {
    __shared__ float E[16][HW];                   // 64 KB

    const int tid  = threadIdx.x;
    const int wave = tid >> 5, lane = tid & 31;
    const int l16 = lane & 15, hi = lane >> 4, kbase = hi * 8;
    const int n0 = blockIdx.x * 16, b = blockIdx.y;

    // preload the 8 A fragments (Fk rows n0..n0+15, K = 0..255) once
    v16bf aF[8];
    const bf16_t* arow = Fk + ((size_t)(b * HW + n0 + l16)) * RR;
    #pragma unroll
    for (int kk = 0; kk < 8; ++kk) aF[kk] = load_frag(arow + kk * 32, kbase);

    for (int mt = wave; mt < 64; mt += 8) {
        int mcol = mt * 16 + l16;
        const bf16_t* brow = Fp + ((size_t)(b * HW + mcol)) * RR;
        v8f acc = v8f{};
        #pragma unroll
        for (int kk = 0; kk < 8; ++kk)
            acc = wmma_bf16(aF[kk], load_frag(brow + kk * 32, kbase), acc);
        #pragma unroll
        for (int v = 0; v < 8; ++v) E[v + hi * 8][mt * 16 + l16] = acc[v];
    }
    __syncthreads();

    // softmax: wave w owns rows 2w, 2w+1
    #pragma unroll
    for (int rr = 0; rr < 2; ++rr) {
        int r = wave * 2 + rr;
        float mx = -3.402823466e38f;
        for (int j = lane; j < HW; j += 32) mx = fmaxf(mx, E[r][j]);
        #pragma unroll
        for (int o = 16; o > 0; o >>= 1) mx = fmaxf(mx, __shfl_xor(mx, o, 32));
        float s = 0.f;
        for (int j = lane; j < HW; j += 32) {
            float e = __expf(E[r][j] - mx);
            E[r][j] = e;
            s += e;
        }
        #pragma unroll
        for (int o = 16; o > 0; o >>= 1) s += __shfl_xor(s, o, 32);
        float inv = 1.f / s;
        bf16_t* srow = S + ((size_t)(b * HW + n0 + r)) * HW;
        for (int j = lane; j < HW; j += 32) srow[j] = f2bf(E[r][j] * inv);
    }
}

// ---------------------------------------------------------------------------
// Kernel 4: out[b,c,n] = x + param * sum_m Fv[c,m] * S[n,m]   (K = HW)
//   Both operands K-contiguous -> direct global fragments. Residual epilogue.
// grid (HW/64, C/64, B), block 256.
// ---------------------------------------------------------------------------
__global__ __launch_bounds__(256) void k_attend(
    const float* __restrict__ x, const bf16_t* __restrict__ Fv,
    const bf16_t* __restrict__ S, const float* __restrict__ param,
    float* __restrict__ out) {
    const int tid  = threadIdx.x;
    const int wave = tid >> 5, lane = tid & 31;
    const int l16 = lane & 15, hi = lane >> 4, kbase = hi * 8;
    const int n0 = blockIdx.x * 64, c0 = blockIdx.y * 64, b = blockIdx.z;

    const int t0 = wave * 2;
    const int row_t = t0 >> 2;
    const int colA  = t0 & 3;

    const bf16_t* arow = Fv + ((size_t)(b * CC + c0 + row_t * 16 + l16)) * HW;
    const bf16_t* brow0 = S + ((size_t)(b * HW + n0 + colA * 16 + l16)) * HW;
    const bf16_t* brow1 = brow0 + (size_t)16 * HW;

    v8f acc[2] = {v8f{}, v8f{}};
    for (int m = 0; m < HW; m += 32) {
        __builtin_prefetch(arow + m + 256, 0, 1);   // global_prefetch_b8
        v16bf aF = load_frag(arow + m, kbase);
        acc[0] = wmma_bf16(aF, load_frag(brow0 + m, kbase), acc[0]);
        acc[1] = wmma_bf16(aF, load_frag(brow1 + m, kbase), acc[1]);
    }

    float p = param[0];
    #pragma unroll
    for (int j = 0; j < 2; ++j) {
        int n = n0 + (colA + j) * 16 + l16;
        #pragma unroll
        for (int v = 0; v < 8; ++v) {
            int c = c0 + row_t * 16 + v + hi * 8;
            size_t idx = ((size_t)(b * CC + c)) * HW + n;
            out[idx] = x[idx] + p * acc[j][v];
        }
    }
}

// ---------------------------------------------------------------------------
// Launch
// ---------------------------------------------------------------------------
extern "C" void kernel_launch(void* const* d_in, const int* in_sizes, int n_in,
                              void* d_out, int out_size, void* d_ws, size_t ws_size,
                              hipStream_t stream) {
    const float* x  = (const float*)d_in[0];
    const float* Wk = (const float*)d_in[1];
    const float* bk = (const float*)d_in[2];
    const float* Wp = (const float*)d_in[3];
    const float* bp = (const float*)d_in[4];
    const float* Wv = (const float*)d_in[5];
    const float* bv = (const float*)d_in[6];
    const float* pm = (const float*)d_in[7];
    float* out = (float*)d_out;

    // workspace layout (bf16 intermediates, all L2-resident: ~135 MB)
    char* ws = (char*)d_ws;
    bf16_t* wWk = (bf16_t*)(ws);                              // 256*512
    bf16_t* wWp = (bf16_t*)(ws + (size_t)262144);             // 256*512
    bf16_t* wWv = (bf16_t*)(ws + (size_t)524288);             // 512*512
    bf16_t* Fk  = (bf16_t*)(ws + (size_t)1048576);            // B*HW*R
    bf16_t* Fp  = (bf16_t*)(ws + (size_t)17825792);           // B*HW*R
    bf16_t* Fv  = (bf16_t*)(ws + (size_t)34603008);           // B*C*HW
    bf16_t* S   = (bf16_t*)(ws + (size_t)68157440);           // B*HW*HW

    k_cvt_bf16<<<dim3((RR * CC + 255) / 256), dim3(256), 0, stream>>>(Wk, wWk, RR * CC);
    k_cvt_bf16<<<dim3((RR * CC + 255) / 256), dim3(256), 0, stream>>>(Wp, wWp, RR * CC);
    k_cvt_bf16<<<dim3((CC * CC + 255) / 256), dim3(256), 0, stream>>>(Wv, wWv, CC * CC);

    k_proj_kp<<<dim3(HW / 64, RR / 64, BB), dim3(256), 0, stream>>>(
        x, wWk, wWp, bk, bp, Fk, Fp);
    k_proj_v<<<dim3(HW / 64, CC / 64, BB), dim3(256), 0, stream>>>(x, wWv, bv, Fv);
    k_energy_softmax<<<dim3(HW / 16, BB), dim3(256), 0, stream>>>(Fk, Fp, S);
    k_attend<<<dim3(HW / 64, CC / 64, BB), dim3(256), 0, stream>>>(x, Fv, S, pm, out);
}
